// CrossGVP_73366631350468
// MI455X (gfx1250) — matrix operand order repaired and checked by reading, not compiled, not run on hardware
//
#include <hip/hip_runtime.h>

#define NL 8192
#define NP 32768
#define EL 262144
#define EP 524288
#define EX 262144
#define HD 96

typedef __attribute__((ext_vector_type(16))) __bf16 bf16x16;
typedef __attribute__((ext_vector_type(16))) short  short16;
typedef __attribute__((ext_vector_type(8)))  float  f32x8;

// ---------------- helpers ----------------
__device__ inline unsigned short f2bf(float f) {
  unsigned int u = __float_as_uint(f);
  u += 0x7fffu + ((u >> 16) & 1u);
  return (unsigned short)(u >> 16);
}

__device__ inline float silu_f(float x) { return x / (1.0f + __expf(-x)); }

__device__ inline void atomicMaxFloat(float* addr, float val) {
  if (val >= 0.0f) atomicMax((int*)addr, __float_as_int(val));
  else             atomicMin((unsigned int*)addr, __float_as_uint(val));
}

// ---------------- conversion kernels ----------------
// A (M x K fp32 row-major) -> Ab (M x Kp bf16, zero padded)
__global__ void cvt_rows(const float* __restrict__ A, unsigned short* __restrict__ B,
                         int M, int K, int Kp) {
  long long i = (long long)blockIdx.x * 256 + threadIdx.x;
  if (i >= (long long)M * Kp) return;
  int m = (int)(i / Kp), kp = (int)(i % Kp);
  B[i] = (kp < K) ? f2bf(A[(long long)m * K + kp]) : (unsigned short)0;
}

// W (K x N fp32 row-major) -> Wt (N x Kp bf16, row n = column n of W, zero padded)
__global__ void cvt_T(const float* __restrict__ W, unsigned short* __restrict__ Wt,
                      int K, int N, int Kp) {
  int i = blockIdx.x * 256 + threadIdx.x;
  if (i >= N * Kp) return;
  int n = i / Kp, kp = i % Kp;
  Wt[i] = (kp < K) ? f2bf(W[kp * N + n]) : (unsigned short)0;
}

// ---------------- WMMA GEMM ----------------
// C[M x Ncols(ldc)] tile = A[M x Kp](bf16) @ Wt^T ; Wt is (N x Kp) bf16.
// epilogue: +bias, optional silu, optional +residual
__device__ inline void load_a_tile(const unsigned short* __restrict__ A, int row, int Kp,
                                   int k0, int lane, short16& out) {
  int kbase = (lane < 16) ? 0 : 8;
#pragma unroll
  for (int j = 0; j < 8; ++j) {
    int k = k0 + kbase + 2 * j + ((j >= 4) ? 8 : 0);
    unsigned int u = *(const unsigned int*)(A + (long long)row * Kp + k);
    out[2 * j]     = (short)(u & 0xFFFFu);
    out[2 * j + 1] = (short)(u >> 16);
  }
}

__device__ inline void load_b_tile(const unsigned short* __restrict__ Wt, int n, int Kp,
                                   int k0, int lane, short16& out) {
  int kbase = (lane < 16) ? 0 : 8;
#pragma unroll
  for (int j = 0; j < 8; ++j) {
    int k = k0 + kbase + 2 * j + ((j >= 4) ? 8 : 0);
    unsigned int u = *(const unsigned int*)(Wt + (long long)n * Kp + k);
    out[2 * j]     = (short)(u & 0xFFFFu);
    out[2 * j + 1] = (short)(u >> 16);
  }
}

__global__ void gemm_wmma(const unsigned short* __restrict__ A,
                          const unsigned short* __restrict__ Wt,
                          const float* __restrict__ bias,
                          const float* __restrict__ residual,
                          float* __restrict__ C,
                          int M, int Kp, int Ntiles, int ldc, int act) {
  int waveId = (int)((blockIdx.x * blockDim.x + threadIdx.x) >> 5);
  int lane = threadIdx.x & 31;
  int total = (M >> 4) * Ntiles;
  if (waveId >= total) return;
  int rowTile = waveId / Ntiles;
  int colTile = waveId % Ntiles;
  int row = rowTile * 16 + (lane & 15);
  int n   = colTile * 16 + (lane & 15);
  f32x8 acc = {0.f, 0.f, 0.f, 0.f, 0.f, 0.f, 0.f, 0.f};
  for (int k0 = 0; k0 < Kp; k0 += 32) {
    short16 ar, br;
    load_a_tile(A, row, Kp, k0, lane, ar);
    load_b_tile(Wt, n, Kp, k0, lane, br);
    bf16x16 a = __builtin_bit_cast(bf16x16, ar);
    bf16x16 b = __builtin_bit_cast(bf16x16, br);
    acc = __builtin_amdgcn_wmma_f32_16x16x32_bf16(false, a, false, b, (short)0, acc,
                                                  false, false);
  }
  float bn = bias ? bias[n] : 0.0f;
#pragma unroll
  for (int r = 0; r < 8; ++r) {
    int mm = rowTile * 16 + r + ((lane < 16) ? 0 : 8);
    float v = acc[r] + bn;
    if (act == 1) v = silu_f(v);
    if (residual) v += residual[(long long)mm * ldc + n];
    C[(long long)mm * ldc + n] = v;
  }
}

// ---------------- GAT edge kernels ----------------
__global__ void init_gat(float* nmax, float* nden, float* agg, int N) {
  int i = blockIdx.x * 256 + threadIdx.x;
  if (i < N) { nmax[i] = -3.0e38f; nden[i] = 0.0f; }
  if (i < N * HD) agg[i] = 0.0f;
}

__global__ void gat_logits(const int* __restrict__ src, const int* __restrict__ tgt,
                           const float* __restrict__ pos,
                           const float* __restrict__ xl, const float* __restrict__ xr,
                           const float* __restrict__ We, const float* __restrict__ att,
                           float* __restrict__ logit, float* __restrict__ nmax, int E) {
  int wid = (int)((blockIdx.x * blockDim.x + threadIdx.x) >> 5);
  int lane = threadIdx.x & 31;
  if (wid >= E) return;
  int s = src[wid], t = tgt[wid];
  float dx = pos[s * 3 + 0] - pos[t * 3 + 0];
  float dy = pos[s * 3 + 1] - pos[t * 3 + 1];
  float dz = pos[s * 3 + 2] - pos[t * 3 + 2];
  float dist = sqrtf(dx * dx + dy * dy + dz * dz);
  const float step = 10.0f / 15.0f;
  const float coeff = -0.5f / (step * step);
  float r[16];
#pragma unroll
  for (int g = 0; g < 16; ++g) { float d = dist - step * g; r[g] = __expf(coeff * d * d); }
  float part = 0.0f;
#pragma unroll
  for (int j = 0; j < 3; ++j) {
    int h = lane + 32 * j;
    float e = xl[(long long)s * HD + h] + xr[(long long)t * HD + h];
#pragma unroll
    for (int g = 0; g < 16; ++g) e += r[g] * We[g * HD + h];
    e = (e > 0.0f) ? e : 0.2f * e;
    part += e * att[h];
  }
#pragma unroll
  for (int off = 16; off > 0; off >>= 1) part += __shfl_down(part, off, 32);
  if (lane == 0) { logit[wid] = part; atomicMaxFloat(&nmax[t], part); }
}

__global__ void gat_expsum(const int* __restrict__ tgt, const float* __restrict__ logit,
                           const float* __restrict__ nmax, float* __restrict__ aexp,
                           float* __restrict__ nden, int E) {
  int e = blockIdx.x * 256 + threadIdx.x;
  if (e >= E) return;
  int t = tgt[e];
  float a = __expf(logit[e] - nmax[t]);
  aexp[e] = a;
  atomicAdd(&nden[t], a);
}

__global__ void gat_scatter(const int* __restrict__ src, const int* __restrict__ tgt,
                            const float* __restrict__ aexp, const float* __restrict__ nden,
                            const float* __restrict__ xl, float* __restrict__ agg, int E) {
  int wid = (int)((blockIdx.x * blockDim.x + threadIdx.x) >> 5);
  int lane = threadIdx.x & 31;
  if (wid >= E) return;
  int s = src[wid], t = tgt[wid];
  float alpha = aexp[wid] / nden[t];
#pragma unroll
  for (int j = 0; j < 3; ++j) {
    int h = lane + 32 * j;
    atomicAdd(&agg[(long long)t * HD + h], alpha * xl[(long long)s * HD + h]);
  }
}

__global__ void gat_update(float* __restrict__ s, const float* __restrict__ agg,
                           const float* __restrict__ bias, int N) {
  int i = blockIdx.x * 256 + threadIdx.x;
  if (i >= N * HD) return;
  int h = i % HD;
  s[i] += silu_f(agg[i] + bias[h]);
}

// ---------------- cross attention ----------------
__global__ void init_gsc(float* g) { g[0] = -3.0e38f; g[1] = 0.0f; }

__global__ void cross_logits(const int* __restrict__ src, const int* __restrict__ tgt,
                             const float* __restrict__ q, const float* __restrict__ k,
                             float* __restrict__ logit, float* __restrict__ gsc, int E) {
  __shared__ float red[8];
  int wid = threadIdx.x >> 5, lane = threadIdx.x & 31;
  int e = blockIdx.x * 8 + wid;
  int valid = (e < E);
  float part = 0.0f;
  if (valid) {
    int s = src[e], t = tgt[e];
#pragma unroll
    for (int j = 0; j < 3; ++j) {
      int h = lane + 32 * j;
      part += q[(long long)t * HD + h] * k[(long long)s * HD + h];
    }
  }
#pragma unroll
  for (int off = 16; off > 0; off >>= 1) part += __shfl_down(part, off, 32);
  float l = part * 0.125f;
  if (lane == 0) { red[wid] = valid ? l : -3.0e38f; if (valid) logit[e] = l; }
  __syncthreads();
  if (threadIdx.x == 0) {
    float m = red[0];
#pragma unroll
    for (int i = 1; i < 8; ++i) m = fmaxf(m, red[i]);
    atomicMaxFloat(&gsc[0], m);
  }
}

__global__ void cross_expsum(const float* __restrict__ logit, float* __restrict__ gsc,
                             float* __restrict__ aexp, int E) {
  __shared__ float red[256];
  int e = blockIdx.x * 256 + threadIdx.x;
  float a = 0.0f;
  if (e < E) { a = __expf(logit[e] - gsc[0]); aexp[e] = a; }
  red[threadIdx.x] = a;
  __syncthreads();
  for (int s = 128; s > 0; s >>= 1) {
    if (threadIdx.x < s) red[threadIdx.x] += red[threadIdx.x + s];
    __syncthreads();
  }
  if (threadIdx.x == 0) atomicAdd(&gsc[1], red[0]);
}

__global__ void cross_scatter(const int* __restrict__ src, const int* __restrict__ tgt,
                              const float* __restrict__ aexp, const float* __restrict__ gsc,
                              const float* __restrict__ v, float* __restrict__ sL, int E) {
  int wid = (int)((blockIdx.x * blockDim.x + threadIdx.x) >> 5);
  int lane = threadIdx.x & 31;
  if (wid >= E) return;
  int s = src[wid], t = tgt[wid];
  float alpha = aexp[wid] / gsc[1];
#pragma unroll
  for (int j = 0; j < 3; ++j) {
    int h = lane + 32 * j;
    atomicAdd(&sL[(long long)t * HD + h], alpha * v[(long long)s * HD + h]);
  }
}

// ---------------- head ----------------
__global__ void head_out(const float* __restrict__ t1, const float* __restrict__ W2,
                         const float* __restrict__ b2, float* __restrict__ out, int M) {
  int i = blockIdx.x * 256 + threadIdx.x;
  if (i >= M * 3) return;
  int m = i / 3, c = i % 3;
  float acc = b2[c];
#pragma unroll
  for (int h = 0; h < HD; ++h) acc += t1[(long long)m * HD + h] * W2[h * 3 + c];
  out[i] = acc;
}

// ---------------- host launcher ----------------
static inline int cdiv(long long a, long long b) { return (int)((a + b - 1) / b); }

extern "C" void kernel_launch(void* const* d_in, const int* in_sizes, int n_in,
                              void* d_out, int out_size, void* d_ws, size_t ws_size,
                              hipStream_t stream) {
  (void)in_sizes; (void)n_in; (void)out_size; (void)ws_size;
  const float* x_L    = (const float*)d_in[0];
  const float* pos_L  = (const float*)d_in[1];
  const float* x_P    = (const float*)d_in[2];
  const float* pos_P  = (const float*)d_in[3];
  const int*   edge_L = (const int*)d_in[4];
  const int*   edge_P = (const int*)d_in[5];
  const int*   c_src  = (const int*)d_in[6];
  const int*   c_tgt  = (const int*)d_in[7];
  const float* Wemb_L = (const float*)d_in[8];  const float* bemb_L = (const float*)d_in[9];
  const float* Wl_L   = (const float*)d_in[10]; const float* bl_L   = (const float*)d_in[11];
  const float* Wr_L   = (const float*)d_in[12]; const float* br_L   = (const float*)d_in[13];
  const float* We_L   = (const float*)d_in[14]; const float* att_L  = (const float*)d_in[15];
  const float* bias_L = (const float*)d_in[16];
  const float* Wemb_P = (const float*)d_in[17]; const float* bemb_P = (const float*)d_in[18];
  const float* Wl_P   = (const float*)d_in[19]; const float* bl_P   = (const float*)d_in[20];
  const float* Wr_P   = (const float*)d_in[21]; const float* br_P   = (const float*)d_in[22];
  const float* We_P   = (const float*)d_in[23]; const float* att_P  = (const float*)d_in[24];
  const float* bias_P = (const float*)d_in[25];
  const float* Win    = (const float*)d_in[26]; const float* b_in   = (const float*)d_in[27];
  const float* Wout   = (const float*)d_in[28]; const float* bout   = (const float*)d_in[29];
  const float* Wq     = (const float*)d_in[30]; const float* bq     = (const float*)d_in[31];
  const float* Wk     = (const float*)d_in[32]; const float* bk     = (const float*)d_in[33];
  const float* Wv     = (const float*)d_in[34]; const float* bv     = (const float*)d_in[35];
  const float* W1     = (const float*)d_in[36]; const float* b1     = (const float*)d_in[37];
  const float* W2     = (const float*)d_in[38]; const float* b2     = (const float*)d_in[39];

  // workspace bump allocator
  char* w = (char*)d_ws;
  auto alloc = [&](size_t bytes) -> void* {
    void* p = (void*)w;
    w += (bytes + 255) & ~(size_t)255;
    return p;
  };
  float* s_L  = (float*)alloc((size_t)NL * HD * 4);
  float* s_P  = (float*)alloc((size_t)NP * HD * 4);
  float* R1   = (float*)alloc((size_t)NP * HD * 4);  // xl / h / q / t1
  float* R2   = (float*)alloc((size_t)NP * HD * 4);  // xr / k
  float* R3   = (float*)alloc((size_t)NP * HD * 4);  // agg / v
  unsigned short* ABF = (unsigned short*)alloc((size_t)NP * HD * 2);
  float* elog = (float*)alloc((size_t)EP * 4);
  float* eexp = (float*)alloc((size_t)EP * 4);
  float* nmax = (float*)alloc((size_t)NP * 4);
  float* nden = (float*)alloc((size_t)NP * 4);
  float* gsc  = (float*)alloc(2 * 4);
  unsigned short* WembLt = (unsigned short*)alloc(96 * 192 * 2);
  unsigned short* WembPt = (unsigned short*)alloc(96 * 32 * 2);
  unsigned short* WlLt   = (unsigned short*)alloc(96 * 96 * 2);
  unsigned short* WrLt   = (unsigned short*)alloc(96 * 96 * 2);
  unsigned short* WlPt   = (unsigned short*)alloc(96 * 96 * 2);
  unsigned short* WrPt   = (unsigned short*)alloc(96 * 96 * 2);
  unsigned short* Wqt    = (unsigned short*)alloc(96 * 96 * 2);
  unsigned short* Wkt    = (unsigned short*)alloc(96 * 96 * 2);
  unsigned short* Wvt    = (unsigned short*)alloc(96 * 96 * 2);
  unsigned short* W1t    = (unsigned short*)alloc(96 * 96 * 2);
  unsigned short* Wint   = (unsigned short*)alloc(384 * 96 * 2);
  unsigned short* Woutt  = (unsigned short*)alloc(96 * 192 * 2);

  auto cvtT = [&](const float* W_, unsigned short* Wt_, int K, int N, int Kp) {
    cvt_T<<<cdiv((long long)N * Kp, 256), 256, 0, stream>>>(W_, Wt_, K, N, Kp);
  };
  auto cvtA = [&](const float* A_, int M, int K, int Kp) {
    cvt_rows<<<cdiv((long long)M * Kp, 256), 256, 0, stream>>>(A_, ABF, M, K, Kp);
  };
  auto gemm = [&](const unsigned short* Wt_, const float* bias_, const float* res_,
                  float* C_, int M, int Kp, int Ntiles, int ldc, int act) {
    long long waves = (long long)(M >> 4) * Ntiles;
    gemm_wmma<<<cdiv(waves, 8), 256, 0, stream>>>(ABF, Wt_, bias_, res_, C_, M, Kp,
                                                  Ntiles, ldc, act);
  };

  // weight conversions
  cvtT(Wemb_L, WembLt, 167, 96, 192);
  cvtT(Wemb_P, WembPt, 21, 96, 32);
  cvtT(Wl_L, WlLt, 96, 96, 96);  cvtT(Wr_L, WrLt, 96, 96, 96);
  cvtT(Wl_P, WlPt, 96, 96, 96);  cvtT(Wr_P, WrPt, 96, 96, 96);
  cvtT(Wq, Wqt, 96, 96, 96);     cvtT(Wk, Wkt, 96, 96, 96);
  cvtT(Wv, Wvt, 96, 96, 96);     cvtT(W1, W1t, 96, 96, 96);
  cvtT(Win, Wint, 96, 384, 96);
  cvtT(Wout, Woutt, 192, 96, 192);

  // ===== Encoder L: s_L = silu(x_L @ Wemb_L + bemb_L) =====
  cvtA(x_L, NL, 167, 192);
  gemm(WembLt, bemb_L, nullptr, s_L, NL, 192, 6, 96, 1);

  // ===== GAT L =====
  cvtA(s_L, NL, 96, 96);
  gemm(WlLt, bl_L, nullptr, R1, NL, 96, 6, 96, 0);  // xl
  gemm(WrLt, br_L, nullptr, R2, NL, 96, 6, 96, 0);  // xr
  init_gat<<<cdiv((long long)NL * HD, 256), 256, 0, stream>>>(nmax, nden, R3, NL);
  gat_logits<<<cdiv(EL, 8), 256, 0, stream>>>(edge_L, edge_L + EL, pos_L, R1, R2, We_L,
                                              att_L, elog, nmax, EL);
  gat_expsum<<<cdiv(EL, 256), 256, 0, stream>>>(edge_L + EL, elog, nmax, eexp, nden, EL);
  gat_scatter<<<cdiv(EL, 8), 256, 0, stream>>>(edge_L, edge_L + EL, eexp, nden, R1, R3, EL);
  gat_update<<<cdiv((long long)NL * HD, 256), 256, 0, stream>>>(s_L, R3, bias_L, NL);

  // ===== Encoder P =====
  cvtA(x_P, NP, 21, 32);
  gemm(WembPt, bemb_P, nullptr, s_P, NP, 32, 6, 96, 1);

  // ===== GAT P =====
  cvtA(s_P, NP, 96, 96);
  gemm(WlPt, bl_P, nullptr, R1, NP, 96, 6, 96, 0);
  gemm(WrPt, br_P, nullptr, R2, NP, 96, 6, 96, 0);
  init_gat<<<cdiv((long long)NP * HD, 256), 256, 0, stream>>>(nmax, nden, R3, NP);
  gat_logits<<<cdiv(EP, 8), 256, 0, stream>>>(edge_P, edge_P + EP, pos_P, R1, R2, We_P,
                                              att_P, elog, nmax, EP);
  gat_expsum<<<cdiv(EP, 256), 256, 0, stream>>>(edge_P + EP, elog, nmax, eexp, nden, EP);
  gat_scatter<<<cdiv(EP, 8), 256, 0, stream>>>(edge_P, edge_P + EP, eexp, nden, R1, R3, EP);
  gat_update<<<cdiv((long long)NP * HD, 256), 256, 0, stream>>>(s_P, R3, bias_P, NP);

  // ===== SSM: h = silu(s_L @ Win + b_in)[:, :192]; s_L += h @ Wout + bout =====
  cvtA(s_L, NL, 96, 96);
  gemm(Wint, b_in, nullptr, R1, NL, 96, 12, 192, 1);   // h: 8192 x 192
  cvtA(R1, NL, 192, 192);
  gemm(Woutt, bout, s_L, s_L, NL, 192, 6, 96, 0);      // s_L = s_L + h@Wout + bout

  // ===== Cross attention =====
  cvtA(s_L, NL, 96, 96);
  gemm(Wqt, bq, nullptr, R1, NL, 96, 6, 96, 0);        // q
  cvtA(s_P, NP, 96, 96);
  gemm(Wkt, bk, nullptr, R2, NP, 96, 6, 96, 0);        // k
  gemm(Wvt, bv, nullptr, R3, NP, 96, 6, 96, 0);        // v
  init_gsc<<<1, 1, 0, stream>>>(gsc);
  cross_logits<<<cdiv(EX, 8), 256, 0, stream>>>(c_src, c_tgt, R1, R2, elog, gsc, EX);
  cross_expsum<<<cdiv(EX, 256), 256, 0, stream>>>(elog, gsc, eexp, EX);
  cross_scatter<<<cdiv(EX, 8), 256, 0, stream>>>(c_src, c_tgt, eexp, gsc, R3, s_L, EX);

  // ===== Head: out = silu(s_L @ W1 + b1) @ W2 + b2 =====
  cvtA(s_L, NL, 96, 96);
  gemm(W1t, b1, nullptr, R1, NL, 96, 6, 96, 1);        // t1
  head_out<<<cdiv((long long)NL * 3, 256), 256, 0, stream>>>(R1, W2, b2, (float*)d_out, NL);
}
